// FusionMamba_25786983646114
// MI455X (gfx1250) — matrix operand order repaired and checked by reading, not compiled, not run on hardware
//
#include <hip/hip_runtime.h>
#include <stdint.h>

// ===========================================================================
// FusionMamba for MI455X (gfx1250, wave32, WMMA).
// All GEMMs: f16 inputs pre-packed into the WMMA 16x16x32 fragment layout,
// v_wmma_f32_16x16x32_f16 compute, f32 accumulate, fused epilogues,
// register double-buffered (software pipelined) fragment loads.
// ===========================================================================

typedef __attribute__((ext_vector_type(16))) _Float16 v16h;
typedef __attribute__((ext_vector_type(8)))  float    v8f;

#define NROWS  16384      // B*T
#define MTILES 1024       // NROWS/16

// ---------------- fragment coordinate helpers (per CDNA5 ISA 7.12.2) -------
// A-matrix 16x32 f16: lane = (m&15) + 16*khalf;  khalf=0 -> K in {0..7,16..23}
//                     khalf=1 -> K in {8..15,24..31};  vgpr v holds K=kk,kk+1
__device__ __forceinline__ int kkA(int lane, int v) {
  return ((v >> 2) << 4) | ((lane >> 4) << 3) | ((v & 3) << 1);
}
// B-matrix 32x16 f16: same K order as A: lanes 0-15 hold K in {0..7,16..23},
// lanes 16-31 hold K in {8..15,24..31}; vgpr v holds K=kk,kk+1 for col=lane&15
__device__ __forceinline__ int kkB(int lane, int v) {
  return ((v >> 2) << 4) | ((lane >> 4) << 3) | ((v & 3) << 1);
}
__device__ __forceinline__ unsigned packpair(float e0, float e1) {
  union { _Float16 h[2]; unsigned u; } p;
  p.h[0] = (_Float16)e0; p.h[1] = (_Float16)e1;
  return p.u;
}

// ---------------- weight packer: f32 W[K,N] -> [ntile][kt][lane][v] --------
__global__ __launch_bounds__(256) void pack_weight(const float* __restrict__ W,
    int K, int N, unsigned* __restrict__ dst, int ktilesP) {
  int tid  = blockIdx.x * 256 + threadIdx.x;
  int v    = tid & 7;
  int lane = (tid >> 3) & 31;
  int f    = tid >> 8;
  int kt   = f % ktilesP;
  int nt   = f / ktilesP;
  int col  = nt * 16 + (lane & 15);
  int k    = kt * 32 + kkB(lane, v);
  float e0 = (col < N && k     < K) ? W[(size_t)k       * N + col] : 0.f;
  float e1 = (col < N && (k+1) < K) ? W[(size_t)(k + 1) * N + col] : 0.f;
  dst[tid] = packpair(e0, e1);
}

// ---------------- activation packer: f32 S[NROWS,lds] -> A-fragments -------
__global__ __launch_bounds__(256) void pack_act(const float* __restrict__ S,
    int lds, int K, unsigned* __restrict__ dst, int ktilesP) {
  int tid  = blockIdx.x * 256 + threadIdx.x;
  int v    = tid & 7;
  int lane = (tid >> 3) & 31;
  int f    = tid >> 8;
  int kt   = f % ktilesP;
  int mt   = f / ktilesP;
  int row  = mt * 16 + (lane & 15);
  int k    = kt * 32 + kkA(lane, v);
  const float* r = S + (size_t)row * lds;
  float e0 = (k     < K) ? r[k]     : 0.f;
  float e1 = (k + 1 < K) ? r[k + 1] : 0.f;
  dst[tid] = packpair(e0, e1);
}

// ---------------- gated pack: ((y + D*xs) * silu(z)) -> A-fragments --------
__global__ __launch_bounds__(256) void pack_gate(const float* __restrict__ y,
    const float* __restrict__ xs, const float* __restrict__ xz,
    const float* __restrict__ Dp, unsigned* __restrict__ dst) {
  int tid  = blockIdx.x * 256 + threadIdx.x;
  int v    = tid & 7;
  int lane = (tid >> 3) & 31;
  int f    = tid >> 8;
  int kt   = f % 25;                 // K=800 -> 25 ktiles exactly
  int mt   = f / 25;
  int row  = mt * 16 + (lane & 15);
  int k    = kt * 32 + kkA(lane, v);
  float e[2];
#pragma unroll
  for (int q = 0; q < 2; ++q) {
    int d = k + q;
    float z = xz[(size_t)row * 1600 + 800 + d];
    float s = 1.f / (1.f + __expf(-z));
    e[q] = (y[(size_t)row * 800 + d] + Dp[d] * xs[(size_t)row * 800 + d]) * z * s;
  }
  dst[tid] = packpair(e[0], e[1]);
}

// ---------------- WMMA GEMM: C[NROWS,N] = epi(A@B + bias) [* aux] ----------
// Block = 256 threads (8 waves) -> 128x128 tile; wave = 2x4 16x16 tiles.
// Fragments fetched straight from fragment-packed global (weights L2-resident)
// with register double-buffering: kt+1 loads issued before kt's 8 WMMAs.
#define EPI_NONE 0
#define EPI_RELU 1
#define EPI_SIG  2
#define EPI_SP   3

template<int EPI, bool HAS_AUX>
__global__ __launch_bounds__(256) void wmma_gemm(
    const unsigned* __restrict__ Ap, const unsigned* __restrict__ Bp,
    const float* __restrict__ bias, const float* __restrict__ aux,
    float* __restrict__ C, int ktiles, int Ncols, int ldc) {
  const int lane   = threadIdx.x & 31;
  const int wave   = threadIdx.x >> 5;
  const int mtile0 = blockIdx.x * 8 + (wave & 3) * 2;
  const int ntile0 = blockIdx.y * 8 + (wave >> 2) * 4;
  const unsigned* a0p = Ap + ((size_t)mtile0 * ktiles) * 256 + lane * 8;
  const unsigned* a1p = a0p + (size_t)ktiles * 256;
  const unsigned* bp  = Bp + ((size_t)ntile0 * ktiles) * 256 + lane * 8;

  v8f acc[2][4];
  const v8f vz = {0.f, 0.f, 0.f, 0.f, 0.f, 0.f, 0.f, 0.f};
#pragma unroll
  for (int i = 0; i < 2; ++i)
#pragma unroll
    for (int j = 0; j < 4; ++j) acc[i][j] = vz;

  // prologue: fragments for kt = 0
  v16h ca0 = *reinterpret_cast<const v16h*>(a0p);
  v16h ca1 = *reinterpret_cast<const v16h*>(a1p);
  v16h cb[4];
#pragma unroll
  for (int j = 0; j < 4; ++j)
    cb[j] = *reinterpret_cast<const v16h*>(bp + (size_t)j * ktiles * 256);

#pragma unroll 2
  for (int kt = 0; kt < ktiles; ++kt) {
    const int nk = (kt + 1 < ktiles) ? kt + 1 : kt;
    // issue next-iteration fragment loads before consuming current ones
    v16h na0 = *reinterpret_cast<const v16h*>(a0p + (size_t)nk * 256);
    v16h na1 = *reinterpret_cast<const v16h*>(a1p + (size_t)nk * 256);
    v16h nb[4];
#pragma unroll
    for (int j = 0; j < 4; ++j)
      nb[j] = *reinterpret_cast<const v16h*>(bp + ((size_t)j * ktiles + nk) * 256);
    if (kt + 2 < ktiles) {           // gfx1250 global_prefetch, distance 2
      __builtin_prefetch(a0p + (size_t)(kt + 2) * 256, 0, 3);
      __builtin_prefetch(a1p + (size_t)(kt + 2) * 256, 0, 3);
    }
#pragma unroll
    for (int j = 0; j < 4; ++j) {
      acc[0][j] = __builtin_amdgcn_wmma_f32_16x16x32_f16(
          false, ca0, false, cb[j], (short)0, acc[0][j], false, false);
      acc[1][j] = __builtin_amdgcn_wmma_f32_16x16x32_f16(
          false, ca1, false, cb[j], (short)0, acc[1][j], false, false);
    }
    ca0 = na0; ca1 = na1;
#pragma unroll
    for (int j = 0; j < 4; ++j) cb[j] = nb[j];
  }

  // C/D layout: lane l -> col = l&15, rows (l>>4)*8 + t  (t = vgpr index)
  const int cn = lane & 15;
  const int rg = (lane >> 4) * 8;
#pragma unroll
  for (int i = 0; i < 2; ++i) {
    int row0 = (mtile0 + i) * 16 + rg;
#pragma unroll
    for (int j = 0; j < 4; ++j) {
      int col = (ntile0 + j) * 16 + cn;
      if (col < Ncols) {
        float bv = bias ? bias[col] : 0.f;
#pragma unroll
        for (int t = 0; t < 8; ++t) {
          float vv = acc[i][j][t] + bv;
          if (EPI == EPI_RELU) vv = fmaxf(vv, 0.f);
          else if (EPI == EPI_SIG) vv = 1.f / (1.f + __expf(-vv));
          else if (EPI == EPI_SP)  vv = (vv > 20.f) ? vv : log1pf(__expf(vv));
          size_t off = (size_t)(row0 + t) * ldc + col;
          if (HAS_AUX) vv *= aux[off];
          C[off] = vv;
        }
      }
    }
  }
}

// ---------------- FiLM: x = audio_vec*gamma + beta -------------------------
__global__ __launch_bounds__(256) void film_kernel(const float* __restrict__ av,
    const float* __restrict__ g, const float* __restrict__ b,
    float* __restrict__ x, int n) {
  int i = blockIdx.x * 256 + threadIdx.x;
  if (i < n) x[i] = av[i] * g[i] + b[i];
}

// ---------------- causal depthwise conv (DC=4) + SiLU ----------------------
__global__ __launch_bounds__(256) void conv_silu(const float* __restrict__ xz,
    const float* __restrict__ cw, const float* __restrict__ cb,
    float* __restrict__ xs) {
  int idx = blockIdx.x * 256 + threadIdx.x;
  if (idx >= NROWS * 800) return;
  int d = idx % 800;
  int n = idx / 800;
  int t = n & 1023;
  float acc = cb[d];
#pragma unroll
  for (int k = 0; k < 4; ++k) {
    int tt = t + k - 3;
    if (tt >= 0) acc += cw[d * 4 + k] * xz[(size_t)(n + k - 3) * 1600 + d];
  }
  xs[idx] = acc / (1.f + __expf(-acc));
}

// ---------------- residual add + LayerNorm (H=400), one wave per row -------
__global__ __launch_bounds__(256) void ln_add(const float* __restrict__ a,
    const float* __restrict__ b, const float* __restrict__ g,
    const float* __restrict__ bb, float* __restrict__ o) {
  int row  = blockIdx.x * 8 + (threadIdx.x >> 5);
  int lane = threadIdx.x & 31;
  const float* ar = a + (size_t)row * 400;
  const float* br = b + (size_t)row * 400;
  float s = 0.f, ss = 0.f;
  for (int i = lane; i < 400; i += 32) {
    float v = ar[i] + br[i]; s += v; ss += v * v;
  }
  for (int off = 16; off; off >>= 1) {
    s  += __shfl_xor(s,  off, 32);
    ss += __shfl_xor(ss, off, 32);
  }
  float mu  = s * (1.f / 400.f);
  float var = ss * (1.f / 400.f) - mu * mu;
  float rs  = rsqrtf(var + 1e-5f);
  float* orow = o + (size_t)row * 400;
  for (int i = lane; i < 400; i += 32) {
    float v = ar[i] + br[i];
    orow[i] = g[i] * (v - mu) * rs + bb[i];
  }
}

// ---------------- selective scan: wave per (b,d), 2 states per lane --------
__global__ __launch_bounds__(256) void scan_kernel(const float* __restrict__ delta,
    const float* __restrict__ xs, const float* __restrict__ dbc,
    const float* __restrict__ A_log, float* __restrict__ y) {
  int wave = threadIdx.x >> 5;
  int lane = threadIdx.x & 31;
  int b    = blockIdx.x / 100;
  int d    = (blockIdx.x % 100) * 8 + wave;
  float A0 = -__expf(A_log[d * 64 + 2 * lane]);
  float A1 = -__expf(A_log[d * 64 + 2 * lane + 1]);
  float h0 = 0.f, h1 = 0.f;
  const size_t rb = (size_t)b * 1024;
  for (int t = 0; t < 1024; ++t) {
    size_t row = rb + t;
    float dv = delta[row * 800 + d];
    float xv = xs[row * 800 + d];
    const float* bc = dbc + row * 160;
    float B0 = bc[25 + 2 * lane], B1 = bc[26 + 2 * lane];
    float C0 = bc[89 + 2 * lane], C1 = bc[90 + 2 * lane];
    float dx = dv * xv;
    h0 = __expf(dv * A0) * h0 + dx * B0;
    h1 = __expf(dv * A1) * h1 + dx * B1;
    float acc = h0 * C0 + h1 * C1;
    for (int off = 16; off; off >>= 1) acc += __shfl_xor(acc, off, 32);
    if (lane == 0) y[row * 800 + d] = acc;
  }
}

// ===========================================================================
extern "C" void kernel_launch(void* const* d_in, const int* in_sizes, int n_in,
                              void* d_out, int out_size, void* d_ws, size_t ws_size,
                              hipStream_t stream) {
  (void)in_sizes; (void)n_in; (void)out_size; (void)ws_size;
  const float* mag_spec      = (const float*)d_in[0];
  const float* audio_enc     = (const float*)d_in[1];
  const float* video_enc     = (const float*)d_in[2];
  const float* audio_proj_w  = (const float*)d_in[3];
  const float* audio_proj_b  = (const float*)d_in[4];
  const float* video_proj_w  = (const float*)d_in[5];
  const float* video_proj_b  = (const float*)d_in[6];
  const float* film_gamma_w  = (const float*)d_in[7];
  const float* film_gamma_b  = (const float*)d_in[8];
  const float* film_beta_w   = (const float*)d_in[9];
  const float* film_beta_b   = (const float*)d_in[10];
  const float* in_proj_w     = (const float*)d_in[11];
  const float* conv_w        = (const float*)d_in[12];
  const float* conv_b        = (const float*)d_in[13];
  const float* x_proj_w      = (const float*)d_in[14];
  const float* dt_proj_w     = (const float*)d_in[15];
  const float* dt_proj_b     = (const float*)d_in[16];
  const float* A_log         = (const float*)d_in[17];
  const float* Dvec          = (const float*)d_in[18];
  const float* out_proj_w    = (const float*)d_in[19];
  const float* mamba_norm_g  = (const float*)d_in[20];
  const float* mamba_norm_b  = (const float*)d_in[21];
  const float* skip_proj_w   = (const float*)d_in[22];
  const float* skip_proj_b   = (const float*)d_in[23];
  const float* fusion_norm_g = (const float*)d_in[24];
  const float* fusion_norm_b = (const float*)d_in[25];
  const float* fc1_w = (const float*)d_in[26];
  const float* fc1_b = (const float*)d_in[27];
  const float* fc2_w = (const float*)d_in[28];
  const float* fc2_b = (const float*)d_in[29];
  const float* fc3_w = (const float*)d_in[30];
  const float* fc3_b = (const float*)d_in[31];
  const float* mag_mask_w = (const float*)d_in[32];
  const float* mag_mask_b = (const float*)d_in[33];
  float* outp = (float*)d_out;

  char* cur = (char*)d_ws;
  auto alloc = [&](size_t bytes) -> void* {
    void* p = (void*)cur;
    cur += (bytes + 255) & ~(size_t)255;
    return p;
  };
  auto frags = [](size_t outer, size_t kt) { return outer * kt * 1024; }; // bytes

  // packed weights  [ntilesP = gn*8][kt][32][8] dwords
  unsigned* w_audio  = (unsigned*)alloc(frags(32, 65));
  unsigned* w_video  = (unsigned*)alloc(frags(32, 16));
  unsigned* w_gamma  = (unsigned*)alloc(frags(32, 13));
  unsigned* w_beta   = (unsigned*)alloc(frags(32, 13));
  unsigned* w_inproj = (unsigned*)alloc(frags(104, 13));
  unsigned* w_xproj  = (unsigned*)alloc(frags(16, 25));
  unsigned* w_dtproj = (unsigned*)alloc(frags(56, 1));
  unsigned* w_outprj = (unsigned*)alloc(frags(32, 25));
  unsigned* w_skip   = (unsigned*)alloc(frags(32, 65));
  unsigned* w_fc1    = (unsigned*)alloc(frags(40, 13));
  unsigned* w_fc2    = (unsigned*)alloc(frags(40, 19));
  unsigned* w_fc3    = (unsigned*)alloc(frags(40, 19));
  unsigned* w_mag    = (unsigned*)alloc(frags(24, 19));
  // packed activations [1024 mtiles][kt][32][8] dwords (with reuse)
  unsigned* p_audio  = (unsigned*)alloc(frags(MTILES, 65));
  unsigned* p_videoe = (unsigned*)alloc(frags(MTILES, 16));
  unsigned* p_vv     = (unsigned*)alloc(frags(MTILES, 13));
  unsigned* p_x      = (unsigned*)alloc(frags(MTILES, 13)); // reused: fused
  unsigned* p_xs     = (unsigned*)alloc(frags(MTILES, 25)); // reused: h1
  unsigned* p_dt     = (unsigned*)alloc(frags(MTILES, 1));
  unsigned* p_y      = (unsigned*)alloc(frags(MTILES, 25)); // reused: h2
  unsigned* p_h3     = (unsigned*)alloc(frags(MTILES, 19));
  unsigned* p_fused = p_x, *p_h1 = p_xs, *p_h2 = p_y;
  // f32 buffers (with reuse)
  float* f_av    = (float*)alloc((size_t)NROWS * 400 * 4);  // reused: fused
  float* f_vv    = (float*)alloc((size_t)NROWS * 400 * 4);  // reused: mamba_out
  float* f_gamma = (float*)alloc((size_t)NROWS * 400 * 4);  // reused: m
  float* f_beta  = (float*)alloc((size_t)NROWS * 400 * 4);  // reused: skip
  float* f_x     = (float*)alloc((size_t)NROWS * 400 * 4);
  float* f_xz    = (float*)alloc((size_t)NROWS * 1600 * 4);
  float* f_xs    = (float*)alloc((size_t)NROWS * 800 * 4);  // reused: h3
  float* f_dbc   = (float*)alloc((size_t)NROWS * 160 * 4);
  float* f_delta = (float*)alloc((size_t)NROWS * 800 * 4);  // reused: h1
  float* f_y     = (float*)alloc((size_t)NROWS * 800 * 4);  // reused: h2
  float* f_m = f_gamma, *f_skip = f_beta, *f_mo = f_vv, *f_fused = f_av;
  float* f_h1 = f_delta, *f_h2 = f_y, *f_h3 = f_xs;

  auto pack_w = [&](const float* W, int K, int N, unsigned* dst) {
    int kt = (K + 31) / 32, ntP = ((N + 127) / 128) * 8;
    pack_weight<<<ntP * kt, 256, 0, stream>>>(W, K, N, dst, kt);
  };
  auto pack_a = [&](const float* S, int lds, int K, unsigned* dst) {
    int kt = (K + 31) / 32;
    pack_act<<<MTILES * kt, 256, 0, stream>>>(S, lds, K, dst, kt);
  };
#define LAUNCH_GEMM(EPI, AUXF, Ap, Bp, biasp, auxp, Cp, K, N, LDC)            \
  do {                                                                        \
    int _kt = ((K) + 31) / 32;                                                \
    dim3 _g(NROWS / 128, ((N) + 127) / 128);                                  \
    wmma_gemm<EPI, AUXF><<<_g, 256, 0, stream>>>(                             \
        (Ap), (Bp), (biasp), (auxp), (Cp), _kt, (N), (LDC));                  \
  } while (0)

  // ---- weight packing (once per call, tiny) ----
  pack_w(audio_proj_w, 2056, 400, w_audio);
  pack_w(video_proj_w, 512, 400, w_video);
  pack_w(film_gamma_w, 400, 400, w_gamma);
  pack_w(film_beta_w,  400, 400, w_beta);
  pack_w(in_proj_w,    400, 1600, w_inproj);
  pack_w(x_proj_w,     800, 153, w_xproj);
  pack_w(dt_proj_w,     25, 800, w_dtproj);
  pack_w(out_proj_w,   800, 400, w_outprj);
  pack_w(skip_proj_w, 2056, 400, w_skip);
  pack_w(fc1_w, 400, 600, w_fc1);
  pack_w(fc2_w, 600, 600, w_fc2);
  pack_w(fc3_w, 600, 600, w_fc3);
  pack_w(mag_mask_w, 600, 257, w_mag);

  // ---- projections + FiLM ----
  pack_a(audio_enc, 2056, 2056, p_audio);
  LAUNCH_GEMM(EPI_NONE, false, p_audio, w_audio, audio_proj_b, nullptr, f_av, 2056, 400, 400);
  pack_a(video_enc, 512, 512, p_videoe);
  LAUNCH_GEMM(EPI_NONE, false, p_videoe, w_video, video_proj_b, nullptr, f_vv, 512, 400, 400);
  pack_a(f_vv, 400, 400, p_vv);
  LAUNCH_GEMM(EPI_NONE, false, p_vv, w_gamma, film_gamma_b, nullptr, f_gamma, 400, 400, 400);
  LAUNCH_GEMM(EPI_NONE, false, p_vv, w_beta, film_beta_b, nullptr, f_beta, 400, 400, 400);
  film_kernel<<<(NROWS * 400 + 255) / 256, 256, 0, stream>>>(f_av, f_gamma, f_beta, f_x, NROWS * 400);

  // ---- mamba ----
  pack_a(f_x, 400, 400, p_x);
  LAUNCH_GEMM(EPI_NONE, false, p_x, w_inproj, nullptr, nullptr, f_xz, 400, 1600, 1600);
  conv_silu<<<(NROWS * 800 + 255) / 256, 256, 0, stream>>>(f_xz, conv_w, conv_b, f_xs);
  pack_a(f_xs, 800, 800, p_xs);
  LAUNCH_GEMM(EPI_NONE, false, p_xs, w_xproj, nullptr, nullptr, f_dbc, 800, 153, 160);
  pack_a(f_dbc, 160, 25, p_dt);
  LAUNCH_GEMM(EPI_SP, false, p_dt, w_dtproj, dt_proj_b, nullptr, f_delta, 25, 800, 800);
  scan_kernel<<<1600, 256, 0, stream>>>(f_delta, f_xs, f_dbc, A_log, f_y);
  pack_gate<<<MTILES * 25, 256, 0, stream>>>(f_y, f_xs, f_xz, Dvec, p_y);
  LAUNCH_GEMM(EPI_NONE, false, p_y, w_outprj, nullptr, nullptr, f_m, 800, 400, 400);

  // ---- residuals / norms / skip ----
  LAUNCH_GEMM(EPI_NONE, false, p_audio, w_skip, skip_proj_b, nullptr, f_skip, 2056, 400, 400);
  ln_add<<<NROWS / 8, 256, 0, stream>>>(f_x, f_m, mamba_norm_g, mamba_norm_b, f_mo);
  ln_add<<<NROWS / 8, 256, 0, stream>>>(f_mo, f_skip, fusion_norm_g, fusion_norm_b, f_fused);

  // ---- MLP head + masked output ----
  pack_a(f_fused, 400, 400, p_fused);
  LAUNCH_GEMM(EPI_RELU, false, p_fused, w_fc1, fc1_b, nullptr, f_h1, 400, 600, 600);
  pack_a(f_h1, 600, 600, p_h1);
  LAUNCH_GEMM(EPI_RELU, false, p_h1, w_fc2, fc2_b, nullptr, f_h2, 600, 600, 600);
  pack_a(f_h2, 600, 600, p_h2);
  LAUNCH_GEMM(EPI_RELU, false, p_h2, w_fc3, fc3_b, nullptr, f_h3, 600, 600, 600);
  pack_a(f_h3, 600, 600, p_h3);
  LAUNCH_GEMM(EPI_SIG, true, p_h3, w_mag, mag_mask_b, mag_spec, outp, 600, 257, 257);
#undef LAUNCH_GEMM
}
